// Contrastive_loss_pairsOnly_71966472012553
// MI455X (gfx1250) — compile-verified, matrix operand order checked by reading, not run on hardware
//
#include <hip/hip_runtime.h>

// CDNA5 / gfx1250: wave32, WMMA 16x16x32 f16 (codegen-confirmed builtin).
typedef __attribute__((ext_vector_type(16))) _Float16 v16h;
typedef __attribute__((ext_vector_type(8)))  float    v8f;

#define MARGIN 0.01f
#define ROW_LEN 65536            // 256*16*16 floats per batch row
#define ROW_V4  (ROW_LEN / 4)    // float4 elements per row
#define BLOCK   256              // 8 wave32s

// 256-thread block sum-reduction using one v_wmma_f32_16x16x32_f16 on wave 0.
// Partials p[0..255] are arranged as A[m][k] with j = m + 16*k (m=0..15,
// k=0..15); K=16..31 are zero. B is all-ones (layout invariant). D[m][n] then
// holds row-sum over K for every n; the 8 D VGPRs per lane are summed
// (lanes 0-15 get rows 0-7, lanes 16-31 get rows 8-15), and a shfl_xor(16)
// folds the two halves. Result valid in all lanes of wave 0.
__device__ __forceinline__ float block_reduce_wmma(float acc, float* lds) {
  const int tid = threadIdx.x;
  lds[tid] = acc;
  __syncthreads();
  float total = 0.0f;
  if (tid < 32) {                       // wave 0: EXEC all-ones for WMMA
    const int m     = tid & 15;
    const int kbase = (tid < 16) ? 0 : 8;   // documented 16-bit A layout
    v16h a;
    #pragma unroll
    for (int i = 0; i < 8; ++i)
      a[i] = (_Float16)lds[(kbase + i) * 16 + m];   // K = kbase..kbase+7
    #pragma unroll
    for (int i = 8; i < 16; ++i)
      a[i] = (_Float16)0.0f;                        // K = 16+kbase.. (unused)
    v16h b;
    #pragma unroll
    for (int i = 0; i < 16; ++i)
      b[i] = (_Float16)1.0f;
    v8f c = {};
    // (neg_a, A, neg_b, B, c_mod, C, reuse_a, reuse_b)
    c = __builtin_amdgcn_wmma_f32_16x16x32_f16(false, a, false, b,
                                               (short)0, c, false, false);
    float s = c[0] + c[1] + c[2] + c[3] + c[4] + c[5] + c[6] + c[7];
    total = s + __shfl_xor(s, 16, 32);
  }
  return total;
}

// One block per batch row: stream 2 x 256 KB with b128 loads, accumulate
// sum((z1-z2)^2) in f32, hinge, write per-row result to workspace.
__global__ __launch_bounds__(BLOCK) void row_mse_hinge_kernel(
    const float4* __restrict__ z1, const float4* __restrict__ z2,
    float* __restrict__ row_out) {
  __shared__ float lds[BLOCK];
  const int  tid  = threadIdx.x;
  const long base = (long)blockIdx.x * ROW_V4;

  float acc = 0.0f;
  #pragma unroll 4
  for (int i = tid; i < ROW_V4; i += BLOCK) {   // 64 float4 pairs per thread
    float4 a = z1[base + i];
    float4 b = z2[base + i];
    float dx = a.x - b.x, dy = a.y - b.y, dz = a.z - b.z, dw = a.w - b.w;
    acc = fmaf(dx, dx, acc);
    acc = fmaf(dy, dy, acc);
    acc = fmaf(dz, dz, acc);
    acc = fmaf(dw, dw, acc);
  }

  float total = block_reduce_wmma(acc, lds);
  if (tid == 0) {
    float mse = total * (1.0f / (float)ROW_LEN);
    row_out[blockIdx.x] = (mse > MARGIN) ? (mse - MARGIN) : 0.0f;
  }
}

// Single block: reduce B per-row hinged values, divide by B, write scalar.
__global__ __launch_bounds__(BLOCK) void final_reduce_kernel(
    const float* __restrict__ rows, float* __restrict__ out, int B) {
  __shared__ float lds[BLOCK];
  const int tid = threadIdx.x;
  float acc = 0.0f;
  for (int i = tid; i < B; i += BLOCK) acc += rows[i];
  float total = block_reduce_wmma(acc, lds);
  if (tid == 0) out[0] = total / (float)B;
}

extern "C" void kernel_launch(void* const* d_in, const int* in_sizes, int n_in,
                              void* d_out, int out_size, void* d_ws, size_t ws_size,
                              hipStream_t stream) {
  const float4* z1 = (const float4*)d_in[0];
  const float4* z2 = (const float4*)d_in[1];
  float* out  = (float*)d_out;
  float* rows = (float*)d_ws;                 // B floats of scratch

  const int B = in_sizes[0] / ROW_LEN;        // 1024 for the reference shapes

  row_mse_hinge_kernel<<<B, BLOCK, 0, stream>>>(z1, z2, rows);
  final_reduce_kernel<<<1, BLOCK, 0, stream>>>(rows, out, B);
}
